// PluginMoodel_35150012351024
// MI455X (gfx1250) — matrix operand order repaired and checked by reading, not compiled, not run on hardware
//
#include <hip/hip_runtime.h>
#include <hip/hip_bf16.h>

// ---------------------------------------------------------------------------
// Types for CDNA5 WMMA
// ---------------------------------------------------------------------------
typedef __attribute__((ext_vector_type(16))) __bf16          v16bf;
typedef __attribute__((ext_vector_type(8)))  float           v8f;
typedef __attribute__((ext_vector_type(8)))  unsigned short  v8us;

union ABf { v8us u[2]; v16bf v; };

enum {
    GF_RELU    = 1,
    GF_ACC     = 2,
    GF_TRANSB  = 4,
    GF_BIASROW = 8,
    GF_BIAS    = 16
};

__device__ __forceinline__ unsigned short f2bf(float f) {
    unsigned int u = __float_as_uint(f);
    u += 0x7FFFu + ((u >> 16) & 1u);          // round-to-nearest-even
    return (unsigned short)(u >> 16);
}
__device__ __forceinline__ unsigned int pk2(float a, float b) {
    return (unsigned)f2bf(a) | ((unsigned)f2bf(b) << 16);
}

// ---------------------------------------------------------------------------
// Generic batched GEMM:  C[M,N] (+)= A[M,K] * B[K,N] (+bias) (ReLU)
// fp32 in global, bf16 in LDS, fp32 accumulate via v_wmma_f32_16x16x32_bf16.
// Block = 256 threads (8 waves), tile 128x128x32, wave grid 4(M) x 2(N),
// each wave owns a 32x64 patch = 2x4 16x16 accumulators (8 WMMA / K-step).
// Uniform fast path for interior blocks (all big GEMMs here are interior).
// ---------------------------------------------------------------------------
__global__ __launch_bounds__(256) void gemm_wmma(
    const float* __restrict__ A, const float* __restrict__ Bm,
    const float* __restrict__ bias, float* __restrict__ Cm,
    int M, int N, int Kd, int ldA, int ldB, int ldC,
    long long sA, long long sB, long long sC, int flags)
{
    __shared__ __align__(16) unsigned short As[128 * 40]; // 128 rows x 32 k (+8 pad)
    __shared__ __align__(16) unsigned short Bs[128 * 40]; // 128 cols x 32 k (+8 pad), k-major

    const int tid  = threadIdx.x;
    const int lane = tid & 31;
    const int wave = tid >> 5;
    const int wm   = (wave >> 1) * 32;   // wave row origin in tile
    const int wn   = (wave & 1) * 64;    // wave col origin in tile
    const int bM   = blockIdx.y * 128;
    const int bN   = blockIdx.x * 128;

    A  += (long long)blockIdx.z * sA;
    Bm += (long long)blockIdx.z * sB;
    Cm += (long long)blockIdx.z * sC;

    const int l15 = lane & 15;
    const int hi  = lane >> 4;           // 0: lanes 0-15, 1: lanes 16-31

    v8f acc[2][4] = {};

    const bool full = (bM + 128 <= M) && (bN + 128 <= N) && ((Kd & 31) == 0);
    const bool trb  = (flags & GF_TRANSB) != 0;

    if (full) {
        // =================== interior fast path (no guards) ===============
        for (int k0 = 0; k0 < Kd; k0 += 32) {
            // A tile: 128x32, float4 per thread x4
            #pragma unroll
            for (int it = 0; it < 4; ++it) {
                int idx = it * 256 + tid;
                int r   = idx >> 3;
                int c4  = (idx & 7) << 2;
                float4 v = *(const float4*)(A + (long long)(bM + r) * ldA + k0 + c4);
                *(uint2*)&As[r * 40 + c4] = make_uint2(pk2(v.x, v.y), pk2(v.z, v.w));
            }
            if (trb) {
                // B stored [N,K]: float4 along k
                #pragma unroll
                for (int it = 0; it < 4; ++it) {
                    int idx = it * 256 + tid;
                    int n   = idx >> 3;
                    int c4  = (idx & 7) << 2;
                    float4 v = *(const float4*)(Bm + (long long)(bN + n) * ldB + k0 + c4);
                    *(uint2*)&Bs[n * 40 + c4] = make_uint2(pk2(v.x, v.y), pk2(v.z, v.w));
                }
            } else {
                // B stored [K,N]: 4 coalesced dword loads down k, packed b64 store
                #pragma unroll
                for (int it = 0; it < 4; ++it) {
                    int idx = it * 256 + tid;
                    int n   = idx & 127;
                    int k4  = (idx >> 7) << 2;
                    const float* p = Bm + (long long)(k0 + k4) * ldB + bN + n;
                    float x0 = p[0];
                    float x1 = p[ldB];
                    float x2 = p[2 * (long long)ldB];
                    float x3 = p[3 * (long long)ldB];
                    *(uint2*)&Bs[n * 40 + k4] = make_uint2(pk2(x0, x1), pk2(x2, x3));
                }
            }
            // prefetch next K-slice into caches (global_prefetch_b8)
            if (k0 + 32 < Kd) {
                __builtin_prefetch(A + (long long)(bM + (tid >> 1)) * ldA +
                                   (k0 + 32) + ((tid & 1) << 4), 0, 3);
                if (trb)
                    __builtin_prefetch(Bm + (long long)(bN + (tid >> 1)) * ldB +
                                       (k0 + 32) + ((tid & 1) << 4), 0, 3);
                else
                    __builtin_prefetch(Bm + (long long)(k0 + 32 + (tid >> 3)) * ldB +
                                       bN + ((tid & 7) << 4), 0, 3);
            }
            __syncthreads();

            ABf afr[2], bfr[4];
            #pragma unroll
            for (int tm = 0; tm < 2; ++tm) {
                const unsigned short* pa = &As[(wm + tm * 16 + l15) * 40 + hi * 8];
                afr[tm].u[0] = *(const v8us*)pa;
                afr[tm].u[1] = *(const v8us*)(pa + 16);
            }
            #pragma unroll
            for (int tn = 0; tn < 4; ++tn) {
                const unsigned short* pb = &Bs[(wn + tn * 16 + l15) * 40 + hi * 8];
                bfr[tn].u[0] = *(const v8us*)pb;
                bfr[tn].u[1] = *(const v8us*)(pb + 16);
            }
            #pragma unroll
            for (int tm = 0; tm < 2; ++tm)
                #pragma unroll
                for (int tn = 0; tn < 4; ++tn)
                    acc[tm][tn] = __builtin_amdgcn_wmma_f32_16x16x32_bf16(
                        false, afr[tm].v, false, bfr[tn].v,
                        (short)0, acc[tm][tn], false, false);
            __syncthreads();
        }
    } else {
        // =================== edge path (scalar guarded loads) =============
        for (int k0 = 0; k0 < Kd; k0 += 32) {
            #pragma unroll
            for (int it = 0; it < 4; ++it) {
                int idx = it * 256 + tid;
                int r   = idx >> 3;
                int c4  = (idx & 7) << 2;
                int gr  = bM + r;
                int gc  = k0 + c4;
                const float* p = A + (long long)gr * ldA + gc;
                bool ok = gr < M;
                float x0 = (ok && gc     < Kd) ? p[0] : 0.f;
                float x1 = (ok && gc + 1 < Kd) ? p[1] : 0.f;
                float x2 = (ok && gc + 2 < Kd) ? p[2] : 0.f;
                float x3 = (ok && gc + 3 < Kd) ? p[3] : 0.f;
                *(uint2*)&As[r * 40 + c4] = make_uint2(pk2(x0, x1), pk2(x2, x3));
            }
            if (trb) {
                #pragma unroll
                for (int it = 0; it < 4; ++it) {
                    int idx = it * 256 + tid;
                    int n   = idx >> 3;
                    int c4  = (idx & 7) << 2;
                    int gn  = bN + n;
                    int gc  = k0 + c4;
                    const float* p = Bm + (long long)gn * ldB + gc;
                    bool ok = gn < N;
                    float x0 = (ok && gc     < Kd) ? p[0] : 0.f;
                    float x1 = (ok && gc + 1 < Kd) ? p[1] : 0.f;
                    float x2 = (ok && gc + 2 < Kd) ? p[2] : 0.f;
                    float x3 = (ok && gc + 3 < Kd) ? p[3] : 0.f;
                    *(uint2*)&Bs[n * 40 + c4] = make_uint2(pk2(x0, x1), pk2(x2, x3));
                }
            } else {
                #pragma unroll
                for (int it = 0; it < 4; ++it) {
                    int idx = it * 256 + tid;
                    int n   = idx & 127;
                    int k4  = (idx >> 7) << 2;
                    int gn  = bN + n;
                    int gk  = k0 + k4;
                    const float* p = Bm + (long long)gk * ldB + gn;
                    bool ok = gn < N;
                    float x0 = (ok && gk     < Kd) ? p[0] : 0.f;
                    float x1 = (ok && gk + 1 < Kd) ? p[ldB] : 0.f;
                    float x2 = (ok && gk + 2 < Kd) ? p[2 * (long long)ldB] : 0.f;
                    float x3 = (ok && gk + 3 < Kd) ? p[3 * (long long)ldB] : 0.f;
                    *(uint2*)&Bs[n * 40 + k4] = make_uint2(pk2(x0, x1), pk2(x2, x3));
                }
            }
            __syncthreads();

            ABf afr[2], bfr[4];
            #pragma unroll
            for (int tm = 0; tm < 2; ++tm) {
                const unsigned short* pa = &As[(wm + tm * 16 + l15) * 40 + hi * 8];
                afr[tm].u[0] = *(const v8us*)pa;
                afr[tm].u[1] = *(const v8us*)(pa + 16);
            }
            #pragma unroll
            for (int tn = 0; tn < 4; ++tn) {
                const unsigned short* pb = &Bs[(wn + tn * 16 + l15) * 40 + hi * 8];
                bfr[tn].u[0] = *(const v8us*)pb;
                bfr[tn].u[1] = *(const v8us*)(pb + 16);
            }
            #pragma unroll
            for (int tm = 0; tm < 2; ++tm)
                #pragma unroll
                for (int tn = 0; tn < 4; ++tn)
                    acc[tm][tn] = __builtin_amdgcn_wmma_f32_16x16x32_bf16(
                        false, afr[tm].v, false, bfr[tn].v,
                        (short)0, acc[tm][tn], false, false);
            __syncthreads();
        }
    }

    // ---- epilogue: VGPR r holds rows r (lanes 0-15) and 8+r (lanes 16-31) ----
    const int rbase = hi * 8;
    #pragma unroll
    for (int tm = 0; tm < 2; ++tm) {
        #pragma unroll
        for (int tn = 0; tn < 4; ++tn) {
            int gcol = bN + wn + tn * 16 + l15;
            if (gcol >= N) continue;
            #pragma unroll
            for (int r = 0; r < 8; ++r) {
                int grow = bM + wm + tm * 16 + rbase + r;
                if (grow >= M) continue;
                float v = acc[tm][tn][r];
                if (flags & GF_BIAS) v += (flags & GF_BIASROW) ? bias[grow] : bias[gcol];
                if (flags & GF_ACC)  v += Cm[(long long)grow * ldC + gcol];
                if (flags & GF_RELU) v = fmaxf(v, 0.f);
                Cm[(long long)grow * ldC + gcol] = v;
            }
        }
    }
}

// ---------------------------------------------------------------------------
// Selector kernels
// ---------------------------------------------------------------------------
__global__ void k_cls(const float* __restrict__ logits, int S, int Kc,
                      int* __restrict__ cls_id)
{
    int b = blockIdx.x;
    __shared__ float sval[256];
    __shared__ int   sidx[256];
    int t = threadIdx.x;
    float mysum = -3.0e38f;
    if (t < Kc) {
        float s = 0.f;
        const float* p = logits + (size_t)b * S * Kc + t;
        for (int ss = 0; ss < S; ++ss) s += p[(size_t)ss * Kc];
        mysum = s;
    }
    sval[t] = mysum; sidx[t] = t;
    __syncthreads();
    for (int o = 128; o > 0; o >>= 1) {
        if (t < o) {
            if (sval[t + o] > sval[t] ||
                (sval[t + o] == sval[t] && sidx[t + o] < sidx[t])) {
                sval[t] = sval[t + o]; sidx[t] = sidx[t + o];
            }
        }
        __syncthreads();
    }
    if (t == 0) cls_id[b] = sidx[0];
}

__global__ void k_scores(const float* __restrict__ logits,
                         const int* __restrict__ cls_id,
                         float* __restrict__ scores, int S, int Kc)
{
    int i = blockIdx.x * blockDim.x + threadIdx.x;
    int b = blockIdx.y;
    if (i >= S) return;
    const float* l = logits + ((size_t)b * S + i) * Kc;
    float mx = -3.0e38f;
    for (int k = 0; k < Kc; ++k) mx = fmaxf(mx, l[k]);
    float sum = 0.f;
    for (int k = 0; k < Kc; ++k) sum += __expf(l[k] - mx);
    scores[(size_t)b * S + i] = __expf(l[cls_id[b]] - mx) / sum;
}

// Deterministic rank-count top-k (stable descending, ties -> lower index)
__global__ void k_rank(const float* __restrict__ scores,
                       int* __restrict__ idxsel, int S, int nsel)
{
    int b = blockIdx.y;
    __shared__ float sc[2304];
    const float* sp = scores + (size_t)b * S;
    for (int i = threadIdx.x; i < S; i += blockDim.x) sc[i] = sp[i];
    __syncthreads();
    int i = blockIdx.x * blockDim.x + threadIdx.x;
    if (i < S) {
        float si = sc[i];
        int r = 0;
        for (int j = 0; j < S; ++j) {
            float sj = sc[j];
            r += (sj > si) || (sj == si && j < i);
        }
        if (r < nsel) idxsel[(size_t)b * nsel + r] = i;
    }
}

__global__ void k_gather(const float* __restrict__ h,
                         const int* __restrict__ idxsel,
                         float* __restrict__ sel,
                         int S, int Fd, int nsel, int toff, int Td)
{
    int pos = blockIdx.x;
    int b   = blockIdx.y;
    int src = idxsel[(size_t)b * nsel + pos];
    const float* sp = h + ((size_t)b * S + src) * Fd;
    float* dp = sel + ((size_t)b * Td + toff + pos) * Fd;
    for (int f = threadIdx.x; f < Fd; f += blockDim.x) dp[f] = sp[f];
}

// ---------------------------------------------------------------------------
// Combiner kernels (pooled/conv kept in [B,J,F] layout)
// ---------------------------------------------------------------------------
__global__ void k_qkvec(const float* __restrict__ qw, const float* __restrict__ qb,
                        const float* __restrict__ kw, const float* __restrict__ kb,
                        float* __restrict__ qv, float* __restrict__ kv,
                        float* __restrict__ qkm, int O, int Fd)
{
    int c = blockIdx.x * blockDim.x + threadIdx.x;
    if (c < Fd) {
        float sq = 0.f, sk = 0.f;
        for (int o = 0; o < O; ++o) {
            sq += qw[(size_t)o * Fd + c];
            sk += kw[(size_t)o * Fd + c];
        }
        qv[c] = sq / (float)O;
        kv[c] = sk / (float)O;
    }
    if (blockIdx.x == 0 && threadIdx.x == 0) {
        float s0 = 0.f, s1 = 0.f;
        for (int o = 0; o < O; ++o) { s0 += qb[o]; s1 += kb[o]; }
        qkm[0] = s0 / (float)O; qkm[1] = s1 / (float)O;
    }
}

__global__ void k_q1k1(const float* __restrict__ pooled,
                       const float* __restrict__ qv, const float* __restrict__ kv,
                       const float* __restrict__ qkm,
                       float* __restrict__ q1, float* __restrict__ k1, int Fd)
{
    int bj = blockIdx.x;
    __shared__ float rq[256], rk[256];
    const float* p = pooled + (size_t)bj * Fd;
    float aq = 0.f, ak = 0.f;
    for (int f = threadIdx.x; f < Fd; f += 256) {
        float v = p[f];
        aq += v * qv[f];
        ak += v * kv[f];
    }
    rq[threadIdx.x] = aq; rk[threadIdx.x] = ak;
    __syncthreads();
    for (int o = 128; o > 0; o >>= 1) {
        if (threadIdx.x < o) {
            rq[threadIdx.x] += rq[threadIdx.x + o];
            rk[threadIdx.x] += rk[threadIdx.x + o];
        }
        __syncthreads();
    }
    if (threadIdx.x == 0) { q1[bj] = rq[0] + qkm[0]; k1[bj] = rk[0] + qkm[1]; }
}

__global__ void k_A1(const float* __restrict__ adj, const float* __restrict__ alpha,
                     const float* __restrict__ q1, const float* __restrict__ k1,
                     float* __restrict__ A1, int J)
{
    int b = blockIdx.y;
    int i = blockIdx.x * blockDim.x + threadIdx.x;
    if (i < J * J) {
        int j = i / J, k = i % J;
        A1[((size_t)b * J + j) * J + k] =
            adj[i] + alpha[0] * tanhf(q1[b * J + j] - k1[b * J + k]);
    }
}

// y[b,k,f] = BN( sum_j A1[b,j,k] * conv[b,j,f] )
__global__ void k_graphbn(const float* __restrict__ conv, const float* __restrict__ A1,
                          const float* __restrict__ bg, const float* __restrict__ bb,
                          const float* __restrict__ bm, const float* __restrict__ bv,
                          float* __restrict__ y, int J, int Fd)
{
    int k = blockIdx.x;
    int b = blockIdx.y;
    __shared__ float a[64];
    if ((int)threadIdx.x < J)
        a[threadIdx.x] = A1[((size_t)b * J + threadIdx.x) * J + k];
    __syncthreads();
    const float* cb = conv + (size_t)b * J * Fd;
    float* yp = y + ((size_t)b * J + k) * Fd;
    for (int f = threadIdx.x; f < Fd; f += blockDim.x) {
        float acc = 0.f;
        for (int j = 0; j < J; ++j) acc += a[j] * cb[(size_t)j * Fd + f];
        acc = (acc - bm[f]) * rsqrtf(bv[f] + 1e-5f) * bg[f] + bb[f];
        yp[f] = acc;
    }
}

__global__ void k_pool1(const float* __restrict__ y, const float* __restrict__ p1w,
                        const float* __restrict__ p1b, float* __restrict__ feat,
                        int J, int Fd)
{
    int b = blockIdx.x;
    for (int f = threadIdx.x; f < Fd; f += blockDim.x) {
        float acc = 0.f;
        for (int j = 0; j < J; ++j) acc += p1w[j] * y[((size_t)b * J + j) * Fd + f];
        feat[(size_t)b * Fd + f] = acc + p1b[0];
    }
}

// ---------------------------------------------------------------------------
// Host orchestration
// ---------------------------------------------------------------------------
static inline void launch_gemm(hipStream_t st, const float* A, const float* Bm,
                               const float* bias, float* Cm,
                               int M, int N, int Kd, int ldA, int ldB, int ldC,
                               long long sA, long long sB, long long sC,
                               int batch, int flags)
{
    dim3 grid((N + 127) / 128, (M + 127) / 128, batch);
    gemm_wmma<<<grid, dim3(256), 0, st>>>(A, Bm, bias, Cm, M, N, Kd,
                                          ldA, ldB, ldC, sA, sB, sC, flags);
}

extern "C" void kernel_launch(void* const* d_in, const int* in_sizes, int n_in,
                              void* d_out, int out_size, void* d_ws, size_t ws_size,
                              hipStream_t stream)
{
    (void)in_sizes; (void)n_in; (void)out_size; (void)ws_size;

    const int Bn = 16, Fd = 1536, Kc = 200, Td = 2720, Jd = 42;
    const int S_[4]  = {2304, 576, 144, 144};
    const int C_[4]  = {192, 384, 768, 1536};
    const int NS_[4] = {2048, 512, 128, 32};
    const int SO_[4] = {0, 2304, 2880, 3024};      // token offsets in hbuf
    const int TO_[4] = {0, 2048, 2560, 2688};      // token offsets in sel

    const float* xin[4] = {(const float*)d_in[0], (const float*)d_in[1],
                           (const float*)d_in[2], (const float*)d_in[3]};

    // ---- workspace layout (all GEMM-visible buffers 16B aligned) ----
    float* ws = (float*)d_ws;
    size_t off = 0;
    float* hbuf   = ws + off; off += (size_t)Bn * 3168 * Fd;  // stage-major [B,S_i,F]
    float* tmp    = ws + off; off += (size_t)Bn * 2304 * Fd;  // MLP hidden
    float* logits = ws + off; off += (size_t)Bn * 2304 * Kc;
    float* sel    = ws + off; off += (size_t)Bn * Td * Fd;    // gathered tokens [B,T,F]
    float* pooled = ws + off; off += (size_t)Bn * Jd * Fd;    // [B,J,F]
    float* convb  = ws + off; off += (size_t)Bn * Jd * Fd;
    float* ybuf   = ws + off; off += (size_t)Bn * Jd * Fd;
    float* feat   = ws + off; off += (size_t)Bn * Fd;
    float* scores = ws + off; off += (size_t)Bn * 2304;
    float* qv     = ws + off; off += Fd;
    float* kv     = ws + off; off += Fd;
    float* q1     = ws + off; off += (size_t)Bn * Jd + 2;     // keep 8B-multiples
    float* k1     = ws + off; off += (size_t)Bn * Jd + 2;
    float* A1     = ws + off; off += (size_t)Bn * Jd * Jd;
    float* qkm    = ws + off; off += 4;
    int*   cls_id = (int*)(ws + off); off += 16;
    int*   idxsel = (int*)(ws + off); off += (size_t)Bn * 2048;

    float* hst[4];
    for (int i = 0; i < 4; ++i) hst[i] = hbuf + (size_t)Bn * SO_[i] * Fd;

    // ---- FPN projections: h_i = relu(x_i @ w1 + b1) @ w2 + b2 ----
    for (int i = 0; i < 4; ++i) {
        const float* w1 = (const float*)d_in[4 + 4 * i];
        const float* b1 = (const float*)d_in[5 + 4 * i];
        const float* w2 = (const float*)d_in[6 + 4 * i];
        const float* b2 = (const float*)d_in[7 + 4 * i];
        launch_gemm(stream, xin[i], w1, b1, tmp, Bn * S_[i], C_[i], C_[i],
                    C_[i], C_[i], C_[i], 0, 0, 0, 1, GF_BIAS | GF_RELU);
        launch_gemm(stream, tmp, w2, b2, hst[i], Bn * S_[i], Fd, C_[i],
                    C_[i], Fd, Fd, 0, 0, 0, 1, GF_BIAS);
    }

    // ---- top-down: h[i-1] += W_up[S_{i-1},S_i] @ h[i] + b (per batch) ----
    for (int i = 3; i >= 1; --i) {
        const float* uw = (const float*)d_in[20 + 2 * (i - 1)];
        const float* ub = (const float*)d_in[21 + 2 * (i - 1)];
        launch_gemm(stream, uw, hst[i], ub, hst[i - 1], S_[i - 1], Fd, S_[i],
                    S_[i], Fd, Fd, 0, (long long)S_[i] * Fd,
                    (long long)S_[i - 1] * Fd, Bn, GF_BIAS | GF_BIASROW | GF_ACC);
    }

    // ---- FPN_UP projections (in place via tmp) ----
    for (int i = 0; i < 4; ++i) {
        const float* w1 = (const float*)d_in[26 + 4 * i];
        const float* b1 = (const float*)d_in[27 + 4 * i];
        const float* w2 = (const float*)d_in[28 + 4 * i];
        const float* b2 = (const float*)d_in[29 + 4 * i];
        launch_gemm(stream, hst[i], w1, b1, tmp, Bn * S_[i], Fd, Fd,
                    Fd, Fd, Fd, 0, 0, 0, 1, GF_BIAS | GF_RELU);
        launch_gemm(stream, tmp, w2, b2, hst[i], Bn * S_[i], Fd, Fd,
                    Fd, Fd, Fd, 0, 0, 0, 1, GF_BIAS);
    }

    // ---- bottom-up: h[i+1] += W_down[S_{i+1},S_i] @ h[i] + b ----
    for (int i = 0; i < 3; ++i) {
        const float* dw = (const float*)d_in[42 + 2 * i];
        const float* db = (const float*)d_in[43 + 2 * i];
        launch_gemm(stream, dw, hst[i], db, hst[i + 1], S_[i + 1], Fd, S_[i],
                    S_[i], Fd, Fd, 0, (long long)S_[i] * Fd,
                    (long long)S_[i + 1] * Fd, Bn, GF_BIAS | GF_BIASROW | GF_ACC);
    }

    // ---- weakly-supervised selector per stage ----
    for (int i = 0; i < 4; ++i) {
        const float* cw = (const float*)d_in[48 + 2 * i];
        const float* cb = (const float*)d_in[49 + 2 * i];
        launch_gemm(stream, hst[i], cw, cb, logits, Bn * S_[i], Kc, Fd,
                    Fd, Kc, Kc, 0, 0, 0, 1, GF_BIAS);
        k_cls<<<dim3(Bn), dim3(256), 0, stream>>>(logits, S_[i], Kc, cls_id);
        k_scores<<<dim3((S_[i] + 255) / 256, Bn), dim3(256), 0, stream>>>(
            logits, cls_id, scores, S_[i], Kc);
        k_rank<<<dim3((S_[i] + 255) / 256, Bn), dim3(256), 0, stream>>>(
            scores, idxsel, S_[i], NS_[i]);
        k_gather<<<dim3(NS_[i], Bn), dim3(256), 0, stream>>>(
            hst[i], idxsel, sel, S_[i], Fd, NS_[i], TO_[i], Td);
    }

    // ---- GCN combiner ----
    const float* p0w = (const float*)d_in[56];
    const float* p0b = (const float*)d_in[57];
    const float* adj = (const float*)d_in[58];
    const float* c1w = (const float*)d_in[59];
    const float* c1b = (const float*)d_in[60];
    const float* qw  = (const float*)d_in[61];
    const float* qb  = (const float*)d_in[62];
    const float* kw  = (const float*)d_in[63];
    const float* kb  = (const float*)d_in[64];
    const float* al  = (const float*)d_in[65];
    const float* bg  = (const float*)d_in[66];
    const float* bb  = (const float*)d_in[67];
    const float* bm  = (const float*)d_in[68];
    const float* bv  = (const float*)d_in[69];
    const float* p1w = (const float*)d_in[70];
    const float* p1b = (const float*)d_in[71];
    const float* fw  = (const float*)d_in[72];
    const float* fb  = (const float*)d_in[73];

    // pooled[b] = p0w[J,T] @ sel[b][T,F] + p0b (per-row bias)
    launch_gemm(stream, p0w, sel, p0b, pooled, Jd, Fd, Td,
                Td, Fd, Fd, 0, (long long)Td * Fd, (long long)Jd * Fd,
                Bn, GF_BIAS | GF_BIASROW);
    // conv[b] = pooled[b][J,F] @ c1w^T + c1b
    launch_gemm(stream, pooled, c1w, c1b, convb, Jd, Fd, Fd,
                Fd, Fd, Fd, (long long)Jd * Fd, 0, (long long)Jd * Fd,
                Bn, GF_BIAS | GF_TRANSB);

    k_qkvec<<<dim3((Fd + 255) / 256), dim3(256), 0, stream>>>(
        qw, qb, kw, kb, qv, kv, qkm, 384, Fd);
    k_q1k1<<<dim3(Bn * Jd), dim3(256), 0, stream>>>(pooled, qv, kv, qkm, q1, k1, Fd);
    k_A1<<<dim3((Jd * Jd + 255) / 256, Bn), dim3(256), 0, stream>>>(
        adj, al, q1, k1, A1, Jd);
    k_graphbn<<<dim3(Jd, Bn), dim3(256), 0, stream>>>(
        convb, A1, bg, bb, bm, bv, ybuf, Jd, Fd);
    k_pool1<<<dim3(Bn), dim3(256), 0, stream>>>(ybuf, p1w, p1b, feat, Jd, Fd);

    // out[b,k] = feat[b,:] @ fw[k,:]^T + fb
    launch_gemm(stream, feat, fw, fb, (float*)d_out, Bn, Kc, Fd,
                Fd, Fd, Kc, 0, 0, 0, 1, GF_BIAS | GF_TRANSB);
}
